// SelfAttention_89386859364426
// MI455X (gfx1250) — compile-verified, matrix operand order checked by reading, not run on hardware
//
#include <hip/hip_runtime.h>

// ---------------------------------------------------------------------------
// Types for CDNA5 WMMA (wave32): v_wmma_f32_16x16x32_bf16
// ---------------------------------------------------------------------------
typedef __attribute__((ext_vector_type(16))) __bf16 v16bf;
typedef __attribute__((ext_vector_type(8)))  __bf16 v8bf;
typedef __attribute__((ext_vector_type(8)))  float  v8f;

#define LATENT   1024
#define LEAF_W   64
#define N_LEAVES 16
#define N_NODES  15
#define NTOK     8192          // B*S = 4*2048
#define SEQ      2048
#define NBATCH   4

__device__ __forceinline__ __bf16 f2bf(float f) {
  unsigned u = __builtin_bit_cast(unsigned, f);
  unsigned r = u + 0x7FFFu + ((u >> 16) & 1u);   // round-to-nearest-even
  unsigned short h = (unsigned short)(r >> 16);
  return __builtin_bit_cast(__bf16, h);
}

// Assemble a 16x16x32 bf16 A/B fragment for one lane.
// Lane layout (CDNA5 ISA, 16-bit A 16x32): lanes 0-15 -> K = kb..kb+7 and
// 16+kb..16+kb+7 with kb=0; lanes 16-31 use kb=8. `row` points at element k0 of
// the lane's matrix row (A: row M, B: row N of the transposed-stored operand).
__device__ __forceinline__ v16bf make_frag(const __bf16* row, int kb) {
  v8bf lo = *(const v8bf*)(row + kb);
  v8bf hi = *(const v8bf*)(row + kb + 16);
  return __builtin_shufflevector(lo, hi, 0,1,2,3,4,5,6,7,8,9,10,11,12,13,14,15);
}

__device__ __forceinline__ v8f wmma_bf16(v16bf a, v16bf b, v8f c) {
  return __builtin_amdgcn_wmma_f32_16x16x32_bf16(false, a, false, b, (short)0, c,
                                                 false, false);
}

// ---------------------------------------------------------------------------
// Prep kernels: bf16 conversion + weight transposes for contiguous B-frags
// ---------------------------------------------------------------------------
__global__ __launch_bounds__(256) void prep_x_kernel(const float* __restrict__ x,
                                                     __bf16* __restrict__ xb) {
  size_t i = (size_t)blockIdx.x * 256 + threadIdx.x;   // 8,388,608 exact
  xb[i] = f2bf(x[i]);
}

// w1: (l, d, h) fp32  ->  w1T: [p][l][h][d] bf16
__global__ __launch_bounds__(256) void prep_w1_kernel(const float* __restrict__ q,
                                                      const float* __restrict__ k,
                                                      const float* __restrict__ v,
                                                      __bf16* __restrict__ w1T) {
  int p = blockIdx.y;
  size_t i = (size_t)blockIdx.x * 256 + threadIdx.x;   // < 1,048,576
  const float* src = (p == 0) ? q : ((p == 1) ? k : v);
  int l = (int)(i >> 16);
  int rem = (int)(i & 65535);
  int d = rem >> 6;
  int h = rem & 63;
  w1T[(((size_t)p * N_LEAVES + l) * LEAF_W + h) * LATENT + d] = f2bf(src[i]);
}

// w2: (l, h, d) fp32  ->  w2T: [p][l][d][h] bf16
__global__ __launch_bounds__(256) void prep_w2_kernel(const float* __restrict__ q,
                                                      const float* __restrict__ k,
                                                      const float* __restrict__ v,
                                                      __bf16* __restrict__ w2T) {
  int p = blockIdx.y;
  size_t i = (size_t)blockIdx.x * 256 + threadIdx.x;   // < 1,048,576
  const float* src = (p == 0) ? q : ((p == 1) ? k : v);
  int l = (int)(i >> 16);
  int rem = (int)(i & 65535);
  int h = rem >> 10;
  int d = rem & 1023;
  w2T[(((size_t)p * N_LEAVES + l) * LATENT + d) * LEAF_W + h] = f2bf(src[i]);
}

__global__ void init_cnt_kernel(int* cnts) {
  if (threadIdx.x < 48) cnts[threadIdx.x] = 0;
}

// ---------------------------------------------------------------------------
// Tree routing: one wave32 per (projection, token). 4 levels of 1024-dot.
// Counting-sort token ids into per-(p,leaf) lists.
// ---------------------------------------------------------------------------
__global__ __launch_bounds__(256)
void fff_route_kernel(const float* __restrict__ x,
                      const float* nwq, const float* nbq,
                      const float* nwk, const float* nbk,
                      const float* nwv, const float* nbv,
                      int* __restrict__ cnts, int* __restrict__ lists) {
  int wave = blockIdx.x * 8 + (threadIdx.x >> 5);  // 24576 waves = 3*8192
  int lane = threadIdx.x & 31;
  int p = wave / NTOK;
  int n = wave - p * NTOK;
  const float* nw = (p == 0) ? nwq : ((p == 1) ? nwk : nwv);
  const float* nb = (p == 0) ? nbq : ((p == 1) ? nbk : nbv);

  const float4* xr = (const float4*)(x + (size_t)n * LATENT);
  float4 xv[8];
#pragma unroll
  for (int c = 0; c < 8; ++c) xv[c] = xr[c * 32 + lane];

  int cur = 0;
#pragma unroll
  for (int lvl = 0; lvl < 4; ++lvl) {
    const float4* wr = (const float4*)(nw + (size_t)cur * LATENT);
    float s = 0.f;
#pragma unroll
    for (int c = 0; c < 8; ++c) {
      float4 wv = wr[c * 32 + lane];
      s += xv[c].x * wv.x + xv[c].y * wv.y + xv[c].z * wv.z + xv[c].w * wv.w;
    }
#pragma unroll
    for (int off = 16; off > 0; off >>= 1) s += __shfl_xor(s, off, 32);
    s += nb[cur];
    cur = 2 * cur + 1 + ((s >= 0.f) ? 1 : 0);
  }
  int leaf = cur - N_NODES;
  if (lane == 0) {
    int pos = atomicAdd(&cnts[p * N_LEAVES + leaf], 1);
    lists[(size_t)(p * N_LEAVES + leaf) * NTOK + pos] = n;
  }
}

// ---------------------------------------------------------------------------
// Per-leaf 2-layer MLP on 16-token tiles. One wave32 per tile.
//   hidden = relu(Xg(16x1024) @ W1[leaf](1024x64) + b1)   [WMMA, K=1024]
//   out    = hidden(16x64)   @ W2[leaf](64x1024) + b2     [WMMA, K=64]
// Scatter out rows into Q/K (row-major bf16) or V^T ([b][d][s] bf16).
// ---------------------------------------------------------------------------
__global__ __launch_bounds__(32)
void fff_mlp_kernel(const __bf16* __restrict__ xb,
                    const __bf16* __restrict__ w1T,
                    const __bf16* __restrict__ w2T,
                    const float* b1q, const float* b1k, const float* b1v,
                    const float* b2q, const float* b2k, const float* b2v,
                    const int* __restrict__ cnts, const int* __restrict__ lists,
                    __bf16* __restrict__ qb, __bf16* __restrict__ kbuf,
                    __bf16* __restrict__ vT) {
  const int p = blockIdx.z, leaf = blockIdx.y, tile = blockIdx.x;
  const int cnt = cnts[p * N_LEAVES + leaf];
  const int row0 = tile * 16;
  if (row0 >= cnt) return;

  const int* lst = lists + (size_t)(p * N_LEAVES + leaf) * NTOK;
  const int lane = threadIdx.x;

  __shared__ __align__(16) __bf16 xs[16][LATENT];
  __shared__ __align__(16) __bf16 hs[16][LEAF_W];
  __shared__ int toks[16];

  if (lane < 16) toks[lane] = (row0 + lane < cnt) ? lst[row0 + lane] : -1;
  __syncthreads();

  for (int r = 0; r < 16; ++r) {
    int tok = toks[r];
    if (tok < 0) tok = toks[0];                      // pad rows: any valid row
    const uint4* src = (const uint4*)(xb + (size_t)tok * LATENT);
    uint4* dst = (uint4*)(&xs[r][0]);
    for (int c = lane; c < 128; c += 32) dst[c] = src[c];
  }
  __syncthreads();

  const int hi = lane >> 4;
  const int li = lane & 15;
  const int kb = hi ? 8 : 0;

  const float* b1 = (p == 0) ? b1q : ((p == 1) ? b1k : b1v);
  const float* b2 = (p == 0) ? b2q : ((p == 1) ? b2k : b2v);
  const __bf16* w1p = w1T + (size_t)(p * N_LEAVES + leaf) * LEAF_W * LATENT;
  const __bf16* w2p = w2T + (size_t)(p * N_LEAVES + leaf) * LATENT * LEAF_W;

  // ---- GEMM 1: 16x64 = (16x1024) x (1024x64) ----
  v8f c0 = {}, c1 = {}, c2 = {}, c3 = {};
  for (int k0 = 0; k0 < LATENT; k0 += 32) {
    v16bf a = make_frag(&xs[li][k0], kb);
    c0 = wmma_bf16(a, make_frag(w1p + (size_t)(0  + li) * LATENT + k0, kb), c0);
    c1 = wmma_bf16(a, make_frag(w1p + (size_t)(16 + li) * LATENT + k0, kb), c1);
    c2 = wmma_bf16(a, make_frag(w1p + (size_t)(32 + li) * LATENT + k0, kb), c2);
    c3 = wmma_bf16(a, make_frag(w1p + (size_t)(48 + li) * LATENT + k0, kb), c3);
  }
#pragma unroll
  for (int r = 0; r < 8; ++r) {
    int m = r + 8 * hi;
    hs[m][0  + li] = f2bf(fmaxf(c0[r] + b1[leaf * LEAF_W + 0  + li], 0.f));
    hs[m][16 + li] = f2bf(fmaxf(c1[r] + b1[leaf * LEAF_W + 16 + li], 0.f));
    hs[m][32 + li] = f2bf(fmaxf(c2[r] + b1[leaf * LEAF_W + 32 + li], 0.f));
    hs[m][48 + li] = f2bf(fmaxf(c3[r] + b1[leaf * LEAF_W + 48 + li], 0.f));
  }
  __syncthreads();

  // ---- GEMM 2: 16x1024 = (16x64) x (64x1024) ----
  v16bf a0 = make_frag(&hs[li][0],  kb);
  v16bf a1 = make_frag(&hs[li][32], kb);
  for (int n0 = 0; n0 < LATENT; n0 += 16) {
    v8f c = {};
    c = wmma_bf16(a0, make_frag(w2p + (size_t)(n0 + li) * LEAF_W + 0,  kb), c);
    c = wmma_bf16(a1, make_frag(w2p + (size_t)(n0 + li) * LEAF_W + 32, kb), c);
#pragma unroll
    for (int r = 0; r < 8; ++r) {
      int m = r + 8 * hi;
      int tok = toks[m];
      if (tok >= 0) {
        float v = c[r] + b2[leaf * LATENT + n0 + li];
        if (p == 0) {
          qb[(size_t)tok * LATENT + n0 + li] = f2bf(v);
        } else if (p == 1) {
          kbuf[(size_t)tok * LATENT + n0 + li] = f2bf(v);
        } else {
          int bidx = tok >> 11, s = tok & (SEQ - 1);
          vT[((size_t)bidx * LATENT + n0 + li) * SEQ + s] = f2bf(v);
        }
      }
    }
  }
}

// ---------------------------------------------------------------------------
// Attention: S = (Q K^T) * 1/32, per batch. One wave per 64x64 block:
// 4x4 register tiling (16 C-frags), 8 fragment loads per 16 WMMAs.
// All 8 waves of a workgroup share the same 64 Q rows (hit WGP cache).
// ---------------------------------------------------------------------------
__global__ __launch_bounds__(256)
void attn_qk_kernel(const __bf16* __restrict__ qb, const __bf16* __restrict__ kbuf,
                    float* __restrict__ S, int batch) {
  int wid = threadIdx.x >> 5, lane = threadIdx.x & 31;
  int m0 = blockIdx.x * 64;
  int n0 = (blockIdx.y * 8 + wid) * 64;
  const __bf16* Q = qb   + (size_t)batch * SEQ * LATENT;
  const __bf16* K = kbuf + (size_t)batch * SEQ * LATENT;
  int hi = lane >> 4, li = lane & 15, kb = hi ? 8 : 0;

  v8f c[4][4];
#pragma unroll
  for (int i = 0; i < 4; ++i)
#pragma unroll
    for (int j = 0; j < 4; ++j) c[i][j] = (v8f){};

  for (int k0 = 0; k0 < LATENT; k0 += 32) {
    v16bf bf[4];
#pragma unroll
    for (int j = 0; j < 4; ++j)
      bf[j] = make_frag(K + (size_t)(n0 + j * 16 + li) * LATENT + k0, kb);
#pragma unroll
    for (int i = 0; i < 4; ++i) {
      v16bf a = make_frag(Q + (size_t)(m0 + i * 16 + li) * LATENT + k0, kb);
#pragma unroll
      for (int j = 0; j < 4; ++j) c[i][j] = wmma_bf16(a, bf[j], c[i][j]);
    }
  }
#pragma unroll
  for (int i = 0; i < 4; ++i)
#pragma unroll
    for (int j = 0; j < 4; ++j)
#pragma unroll
      for (int r = 0; r < 8; ++r) {
        int m = m0 + i * 16 + r + 8 * hi;
        S[(size_t)m * SEQ + n0 + j * 16 + li] = c[i][j][r] * 0.03125f;
      }
}

__global__ __launch_bounds__(256)
void softmax_kernel(const float* __restrict__ S, __bf16* __restrict__ P) {
  int row = blockIdx.x;
  const float* s = S + (size_t)row * SEQ;
  __shared__ float red[256];
  int t = threadIdx.x;
  float mx = -3.4e38f;
  for (int j = t; j < SEQ; j += 256) mx = fmaxf(mx, s[j]);
  red[t] = mx;
  __syncthreads();
  for (int o = 128; o > 0; o >>= 1) {
    if (t < o) red[t] = fmaxf(red[t], red[t + o]);
    __syncthreads();
  }
  mx = red[0];
  __syncthreads();
  float sum = 0.f;
  for (int j = t; j < SEQ; j += 256) sum += __expf(s[j] - mx);
  red[t] = sum;
  __syncthreads();
  for (int o = 128; o > 0; o >>= 1) {
    if (t < o) red[t] += red[t + o];
    __syncthreads();
  }
  float inv = 1.f / red[0];
  __bf16* p = P + (size_t)row * SEQ;
  for (int j = t; j < SEQ; j += 256) p[j] = f2bf(__expf(s[j] - mx) * inv);
}

// O = P @ V  (A = P row-major bf16 stride 2048, B frags from V^T contiguous).
// Same 4x4 register tiling, 64x64 output block per wave, K = 2048.
__global__ __launch_bounds__(256)
void attn_pv_kernel(const __bf16* __restrict__ P, const __bf16* __restrict__ vT,
                    float* __restrict__ out, int batch) {
  int wid = threadIdx.x >> 5, lane = threadIdx.x & 31;
  int m0 = blockIdx.x * 64;
  int n0 = (blockIdx.y * 8 + wid) * 64;
  const __bf16* V = vT + (size_t)batch * LATENT * SEQ;
  int hi = lane >> 4, li = lane & 15, kb = hi ? 8 : 0;

  v8f c[4][4];
#pragma unroll
  for (int i = 0; i < 4; ++i)
#pragma unroll
    for (int j = 0; j < 4; ++j) c[i][j] = (v8f){};

  for (int k0 = 0; k0 < SEQ; k0 += 32) {
    v16bf bf[4];
#pragma unroll
    for (int j = 0; j < 4; ++j)
      bf[j] = make_frag(V + (size_t)(n0 + j * 16 + li) * SEQ + k0, kb);
#pragma unroll
    for (int i = 0; i < 4; ++i) {
      v16bf a = make_frag(P + (size_t)(m0 + i * 16 + li) * SEQ + k0, kb);
#pragma unroll
      for (int j = 0; j < 4; ++j) c[i][j] = wmma_bf16(a, bf[j], c[i][j]);
    }
  }
  float* o = out + (size_t)batch * SEQ * LATENT;
#pragma unroll
  for (int i = 0; i < 4; ++i)
#pragma unroll
    for (int j = 0; j < 4; ++j)
#pragma unroll
      for (int r = 0; r < 8; ++r) {
        int m = m0 + i * 16 + r + 8 * hi;
        o[(size_t)m * LATENT + n0 + j * 16 + li] = c[i][j][r];
      }
}

// ---------------------------------------------------------------------------
// Host launcher
// ---------------------------------------------------------------------------
extern "C" void kernel_launch(void* const* d_in, const int* in_sizes, int n_in,
                              void* d_out, int out_size, void* d_ws, size_t ws_size,
                              hipStream_t stream) {
  (void)in_sizes; (void)n_in; (void)out_size; (void)ws_size;

  const float* x = (const float*)d_in[0];
  const float* nw[3]  = {(const float*)d_in[1],  (const float*)d_in[7],  (const float*)d_in[13]};
  const float* nb[3]  = {(const float*)d_in[2],  (const float*)d_in[8],  (const float*)d_in[14]};
  const float* w1[3]  = {(const float*)d_in[3],  (const float*)d_in[9],  (const float*)d_in[15]};
  const float* b1[3]  = {(const float*)d_in[4],  (const float*)d_in[10], (const float*)d_in[16]};
  const float* w2[3]  = {(const float*)d_in[5],  (const float*)d_in[11], (const float*)d_in[17]};
  const float* b2[3]  = {(const float*)d_in[6],  (const float*)d_in[12], (const float*)d_in[18]};

  char* ws = (char*)d_ws;
  size_t off = 0;
  auto take = [&](size_t bytes) -> char* {
    char* p = ws + off;
    off += (bytes + 255) & ~(size_t)255;
    return p;
  };
  __bf16* xb   = (__bf16*)take((size_t)NTOK * LATENT * 2);
  __bf16* w1T  = (__bf16*)take((size_t)3 * N_LEAVES * LEAF_W * LATENT * 2);
  __bf16* w2T  = (__bf16*)take((size_t)3 * N_LEAVES * LATENT * LEAF_W * 2);
  __bf16* qb   = (__bf16*)take((size_t)NTOK * LATENT * 2);
  __bf16* kbuf = (__bf16*)take((size_t)NTOK * LATENT * 2);
  __bf16* vT   = (__bf16*)take((size_t)NBATCH * LATENT * SEQ * 2);
  float*  S    = (float*)take((size_t)SEQ * SEQ * 4);       // reused per batch
  __bf16* P    = (__bf16*)take((size_t)SEQ * SEQ * 2);      // reused per batch
  int*    cnts = (int*)take(48 * 4);
  int*    lists= (int*)take((size_t)3 * N_LEAVES * NTOK * 4);

  // Prep: bf16 conversions + weight transposes
  prep_x_kernel<<<32768, 256, 0, stream>>>(x, xb);
  prep_w1_kernel<<<dim3(4096, 3), 256, 0, stream>>>(w1[0], w1[1], w1[2], w1T);
  prep_w2_kernel<<<dim3(4096, 3), 256, 0, stream>>>(w2[0], w2[1], w2[2], w2T);
  init_cnt_kernel<<<1, 64, 0, stream>>>(cnts);

  // Tree routing + counting sort
  fff_route_kernel<<<3072, 256, 0, stream>>>(x, nw[0], nb[0], nw[1], nb[1],
                                             nw[2], nb[2], cnts, lists);

  // Per-leaf MLPs (WMMA)
  fff_mlp_kernel<<<dim3(NTOK / 16, N_LEAVES, 3), 32, 0, stream>>>(
      xb, w1T, w2T, b1[0], b1[1], b1[2], b2[0], b2[1], b2[2],
      cnts, lists, qb, kbuf, vT);

  // Attention, one batch at a time (S/P scratch reused; stream order serializes)
  float* out = (float*)d_out;
  for (int b = 0; b < NBATCH; ++b) {
    attn_qk_kernel<<<dim3(SEQ / 64, 4), 256, 0, stream>>>(qb, kbuf, S, b);
    softmax_kernel<<<SEQ, 256, 0, stream>>>(S, P);
    attn_pv_kernel<<<dim3(SEQ / 64, 2), 256, 0, stream>>>(P, vT, out, b);
  }
}
